// Bidirect_LSTM_Max_Pooling_46651934769705
// MI455X (gfx1250) — matrix operand order repaired and checked by reading, not compiled, not run on hardware
//
#include <hip/hip_runtime.h>
#include <hip/hip_bf16.h>
#include <float.h>

typedef unsigned short u16;
typedef __bf16 bf16_t;
typedef bf16_t v16bf __attribute__((ext_vector_type(16)));
typedef float  v8f   __attribute__((ext_vector_type(8)));

#define B_  64
#define S_  512
#define E_  300
#define EP  320      // K padded to multiple of 32
#define H_  512
#define G3  1536     // packed gate cols: i | g | o  (f-gate is dead: c_prev == 0)
#define LDS_STRIDE 520   // 512 + 8 u16 pad per row (bank-conflict avoidance)

union FragU { uint4 q[2]; v16bf v; };

__device__ inline u16 f2bf(float f) {
  unsigned u = __float_as_uint(f);
  unsigned r = u + 0x7FFFu + ((u >> 16) & 1u);   // RNE
  return (u16)(r >> 16);
}
__device__ inline float bf2f(u16 h) { return __uint_as_float(((unsigned)h) << 16); }

// v_rcp_f32-based activations: 1 exp + 1 rcp each, no IEEE divide, no clamp
// (exp(-x) overflow -> rcp(inf) = 0 gives the right saturation).
__device__ inline float fast_sigmoid(float x) {
  return __builtin_amdgcn_rcpf(1.0f + __expf(-x));
}
__device__ inline float fast_tanh(float x) {
  // tanh(x) = 2*sigmoid(2x) - 1
  return 2.0f * __builtin_amdgcn_rcpf(1.0f + __expf(-2.0f * x)) - 1.0f;
}

__device__ inline v8f vzero() {
  v8f z;
#pragma unroll
  for (int i = 0; i < 8; ++i) z[i] = 0.0f;
  return z;
}
__device__ inline v8f vneginf() {
  v8f z;
#pragma unroll
  for (int i = 0; i < 8; ++i) z[i] = -FLT_MAX;
  return z;
}

// A fragment: 16x32 bf16 tile from row-major [M, ld], rows m0.., K cols kb..kb+31.
__device__ inline v16bf load_a_frag(const u16* base, int ld, int m0, int kb, int lane) {
  int hl = lane >> 4, lr = lane & 15;
  const u16* p = base + (size_t)(m0 + lr) * ld + kb + hl * 16;
  FragU f;
  f.q[0] = *(const uint4*)p;         // K + {0..7}/{16..23}
  f.q[1] = *(const uint4*)(p + 8);   // K + {8..15}/{24..31}
  return f.v;
}
// B fragment for gates = A @ W^T: B[k,n] = W[n,k] with W row-major [N, ld].
__device__ inline v16bf load_b_frag(const u16* base, int ld, int n0, int kb, int lane) {
  int hl = lane >> 4, lr = lane & 15;
  const u16* p = base + (size_t)(n0 + lr) * ld + kb + hl * 16;
  FragU f;
  f.q[0] = *(const uint4*)p;
  f.q[1] = *(const uint4*)(p + 8);
  return f.v;
}
__device__ inline v16bf load_b_lds(const u16* lds, int n0, int kb, int lane) {
  int hl = lane >> 4, lr = lane & 15;
  const u16* p = lds + (n0 + lr) * LDS_STRIDE + kb + hl * 16;
  FragU f;
  f.q[0] = *(const uint4*)p;
  f.q[1] = *(const uint4*)(p + 8);
  return f.v;
}

__device__ inline v8f wmma_bf16(v16bf a, v16bf b, v8f c) {
  return __builtin_amdgcn_wmma_f32_16x16x32_bf16(false, a, false, b, (short)0, c, false, false);
}

// ---------------- prologue kernels ----------------

__global__ void init_kernel(u16* hping, u16* hpong, unsigned* bar) {
  int i = blockIdx.x * blockDim.x + threadIdx.x;
  if (i < B_ * H_) { hping[i] = 0; hpong[i] = 0; }
  if (i < 8) bar[i] = 0;
}

__global__ void embed_kernel(const int* __restrict__ idx, const float* __restrict__ table,
                             u16* __restrict__ xbf) {
  int row = blockIdx.x;                       // (b,s) flattened, 0..32767
  int tok = idx[row];
  const float* src = table + (size_t)tok * E_;
  u16* dst = xbf + (size_t)row * EP;
  for (int c = threadIdx.x; c < EP; c += blockDim.x) {
    float v = (c < E_) ? src[c] : 0.0f;
    dst[c] = f2bf(v);
  }
}

// Repack i/g/o rows (drop f), pad K 300->320, bf16; also fuse biases.
__global__ void pack_kernel(const float* __restrict__ Wihf, const float* __restrict__ Wihb,
                            const float* __restrict__ Whhb,
                            const float* __restrict__ bihf, const float* __restrict__ bhhf,
                            const float* __restrict__ bihb, const float* __restrict__ bhhb,
                            u16* __restrict__ Wf, u16* __restrict__ Wb, u16* __restrict__ Whh,
                            float* __restrict__ biasF, float* __restrict__ biasB) {
  int p = blockIdx.x;                 // packed row 0..1535
  int gate = p >> 9, w = p & 511;
  int src = (gate == 0 ? 0 : (gate == 1 ? 2 * H_ : 3 * H_)) + w;   // i, g, o rows of [4H, .]
  for (int c = threadIdx.x; c < EP; c += blockDim.x) {
    float vf = (c < E_) ? Wihf[(size_t)src * E_ + c] : 0.0f;
    float vb = (c < E_) ? Wihb[(size_t)src * E_ + c] : 0.0f;
    Wf[(size_t)p * EP + c] = f2bf(vf);
    Wb[(size_t)p * EP + c] = f2bf(vb);
  }
  for (int c = threadIdx.x; c < H_; c += blockDim.x)
    Whh[(size_t)p * H_ + c] = f2bf(Whhb[(size_t)src * H_ + c]);
  if (threadIdx.x == 0) {
    biasF[p] = bihf[src] + bhhf[src];
    biasB[p] = bihb[src] + bhhb[src];
  }
}

// ---------------- forward: gates GEMM + cell + max over S, fused ----------------
// Grid = 64 b x 4 n-groups = 256 WGs. Each wave owns one 16-col h tile of one b,
// loops all 32 M-tiles (s) keeping the time-max in registers. No atomics.
__global__ __launch_bounds__(256) void fwd_kernel(const u16* __restrict__ xbf,
                                                  const u16* __restrict__ Wf,
                                                  const float* __restrict__ biasF,
                                                  float* __restrict__ out) {
  int b = blockIdx.x & 63;
  int ng = blockIdx.x >> 6;                   // 0..3
  int wave = threadIdx.x >> 5, lane = threadIdx.x & 31;
  int hl = lane >> 4, lr = lane & 15;
  const u16* xrow = xbf + (size_t)b * S_ * EP;
  int n0 = (ng * 8 + wave) * 16;              // h-col base, 0..496
  float bi_ = biasF[n0 + lr], bg_ = biasF[512 + n0 + lr], bo_ = biasF[1024 + n0 + lr];
  v8f mx = vneginf();
  for (int mt = 0; mt < 32; ++mt) {
    int m0 = mt * 16;
    v8f ai = vzero(), ag = vzero(), ao = vzero();
#pragma unroll 2
    for (int kc = 0; kc < EP / 32; ++kc) {
      int kb = kc * 32;
      v16bf a  = load_a_frag(xrow, EP, m0, kb, lane);
      v16bf wi = load_b_frag(Wf, EP, n0, kb, lane);
      v16bf wg = load_b_frag(Wf, EP, 512 + n0, kb, lane);
      v16bf wo = load_b_frag(Wf, EP, 1024 + n0, kb, lane);
      ai = wmma_bf16(a, wi, ai);
      ag = wmma_bf16(a, wg, ag);
      ao = wmma_bf16(a, wo, ao);
    }
#pragma unroll
    for (int r = 0; r < 8; ++r) {
      float c = fast_sigmoid(ai[r] + bi_) * fast_tanh(ag[r] + bg_);
      float h = fast_sigmoid(ao[r] + bo_) * fast_tanh(c);
      mx[r] = fmaxf(mx[r], h);
    }
  }
  float cm = mx[0];
#pragma unroll
  for (int r = 1; r < 8; ++r) cm = fmaxf(cm, mx[r]);
  cm = fmaxf(cm, __shfl_xor(cm, 16, 32));     // combine row-halves (wave32)
  if (lane < 16) out[(size_t)b * 1024 + n0 + lane] = cm;
}

// ---------------- backward input-gates GEMM ----------------
// xg layout: [S][1536][64] bf16 (col-major per step) so the scan's C-addends are
// one contiguous b128 per lane. Bias folded here.
__global__ __launch_bounds__(256) void bgates_kernel(const u16* __restrict__ xbf,
                                                     const u16* __restrict__ Wb,
                                                     const float* __restrict__ biasB,
                                                     u16* __restrict__ xg) {
  int b = blockIdx.x & 63;
  int ng = blockIdx.x >> 6;
  int wave = threadIdx.x >> 5, lane = threadIdx.x & 31;
  int hl = lane >> 4, lr = lane & 15;
  const u16* xrow = xbf + (size_t)b * S_ * EP;
  int n0 = (ng * 8 + wave) * 16;
  float bi_ = biasB[n0 + lr], bg_ = biasB[512 + n0 + lr], bo_ = biasB[1024 + n0 + lr];
  for (int mt = 0; mt < 32; ++mt) {
    int m0 = mt * 16;
    v8f ai = vzero(), ag = vzero(), ao = vzero();
#pragma unroll 2
    for (int kc = 0; kc < EP / 32; ++kc) {
      int kb = kc * 32;
      v16bf a  = load_a_frag(xrow, EP, m0, kb, lane);
      v16bf wi = load_b_frag(Wb, EP, n0, kb, lane);
      v16bf wg = load_b_frag(Wb, EP, 512 + n0, kb, lane);
      v16bf wo = load_b_frag(Wb, EP, 1024 + n0, kb, lane);
      ai = wmma_bf16(a, wi, ai);
      ag = wmma_bf16(a, wg, ag);
      ao = wmma_bf16(a, wo, ao);
    }
#pragma unroll
    for (int r = 0; r < 8; ++r) {
      int s = m0 + hl * 8 + r;
      u16* dst = xg + (size_t)s * G3 * B_;
      dst[(size_t)(n0 + lr) * B_ + b]          = f2bf(ai[r] + bi_);
      dst[(size_t)(512 + n0 + lr) * B_ + b]    = f2bf(ag[r] + bg_);
      dst[(size_t)(1024 + n0 + lr) * B_ + b]   = f2bf(ao[r] + bo_);
    }
  }
}

// ---------------- persistent backward scan: 16 WGs, grid barrier per step ----------------
__device__ inline void grid_barrier(unsigned* cnt, unsigned* gen, unsigned nwg) {
  __threadfence();           // release h stores device-wide
  __syncthreads();
  if (threadIdx.x == 0) {
    unsigned g = __hip_atomic_load(gen, __ATOMIC_RELAXED, __HIP_MEMORY_SCOPE_AGENT);
    unsigned a = __hip_atomic_fetch_add(cnt, 1u, __ATOMIC_ACQ_REL, __HIP_MEMORY_SCOPE_AGENT);
    if (a == nwg - 1) {
      __hip_atomic_store(cnt, 0u, __ATOMIC_RELAXED, __HIP_MEMORY_SCOPE_AGENT);
      __hip_atomic_fetch_add(gen, 1u, __ATOMIC_RELEASE, __HIP_MEMORY_SCOPE_AGENT);
    } else {
      while (__hip_atomic_load(gen, __ATOMIC_ACQUIRE, __HIP_MEMORY_SCOPE_AGENT) == g)
        __builtin_amdgcn_s_sleep(4);
    }
  }
  __syncthreads();
  __threadfence();           // acquire other WGs' h stores
}

__global__ __launch_bounds__(256) void scan_kernel(const u16* __restrict__ Whh,
                                                   const u16* __restrict__ xg,
                                                   u16* __restrict__ hping, u16* __restrict__ hpong,
                                                   unsigned* __restrict__ bar,
                                                   float* __restrict__ out) {
  extern __shared__ u16 ldsW[];                  // 96 rows (32 i | 32 g | 32 o) x 512, stride 520
  int wg = blockIdx.x;                           // 0..15; owns h cols [wg*32, wg*32+32)
  int c0 = wg * 32;
  // Stage this WG's W_hh rows into LDS once (96 KB, uint4 copies)
  for (int i = threadIdx.x; i < 96 * (H_ / 8); i += blockDim.x) {
    int row = i >> 6, seg = i & 63;              // 64 x 16B per row
    int gate = row >> 5, rr = row & 31;
    const uint4* src = (const uint4*)(Whh + (size_t)(gate * 512 + c0 + rr) * H_) + seg;
    *(uint4*)(&ldsW[row * LDS_STRIDE + seg * 8]) = *src;
  }
  __syncthreads();

  int wave = threadIdx.x >> 5, lane = threadIdx.x & 31;
  int hl = lane >> 4, lr = lane & 15;
  int m0 = (wave & 3) * 16;                      // batch-row tile
  int nblk = (wave >> 2) * 16;                   // 0 or 16 within this WG's 32 cols
  int hcol0 = c0 + nblk;
  int boff = m0 + hl * 8;

  v8f mx = vneginf();
  const u16* hin = hping;
  u16* hout = hpong;

  for (int t = 0; t < S_; ++t) {
    int tsrc = S_ - 1 - t;                       // backward LSTM: reversed time
    v8f ai = vzero(), ag = vzero(), ao = vzero();
#pragma unroll 4
    for (int kc = 0; kc < H_ / 32; ++kc) {
      int kb = kc * 32;
      v16bf a  = load_a_frag(hin, H_, m0, kb, lane);
      v16bf wi = load_b_lds(ldsW, 0  + nblk, kb, lane);
      v16bf wg = load_b_lds(ldsW, 32 + nblk, kb, lane);
      v16bf wo = load_b_lds(ldsW, 64 + nblk, kb, lane);
      ai = wmma_bf16(a, wi, ai);
      ag = wmma_bf16(a, wg, ag);
      ao = wmma_bf16(a, wo, ao);
    }
    // xg addends: one contiguous b128 per gate per lane ([S][1536][64] layout)
    const u16* xt = xg + (size_t)tsrc * G3 * B_;
    union { uint4 q; u16 h[8]; } xi, xgv, xo;
    xi.q  = *(const uint4*)(xt + (size_t)(hcol0 + lr) * B_ + boff);
    xgv.q = *(const uint4*)(xt + (size_t)(512 + hcol0 + lr) * B_ + boff);
    xo.q  = *(const uint4*)(xt + (size_t)(1024 + hcol0 + lr) * B_ + boff);
#pragma unroll
    for (int r = 0; r < 8; ++r) {
      float vi = ai[r] + bf2f(xi.h[r]);
      float vg = ag[r] + bf2f(xgv.h[r]);
      float vo = ao[r] + bf2f(xo.h[r]);
      float c = fast_sigmoid(vi) * fast_tanh(vg);
      float h = fast_sigmoid(vo) * fast_tanh(c);
      mx[r] = fmaxf(mx[r], h);
      hout[(size_t)(m0 + hl * 8 + r) * H_ + hcol0 + lr] = f2bf(h);
    }
    grid_barrier(bar, bar + 1, 16);
    const u16* th = hin; hin = hout; hout = (u16*)th;   // ping-pong
  }
#pragma unroll
  for (int r = 0; r < 8; ++r)
    out[(size_t)(m0 + hl * 8 + r) * 1024 + 512 + hcol0 + lr] = mx[r];
}

// ---------------- launch ----------------
extern "C" void kernel_launch(void* const* d_in, const int* in_sizes, int n_in,
                              void* d_out, int out_size, void* d_ws, size_t ws_size,
                              hipStream_t stream) {
  (void)in_sizes; (void)n_in; (void)out_size; (void)ws_size;
  const int*   idx   = (const int*)d_in[0];
  const float* table = (const float*)d_in[1];
  const float* Wihf  = (const float*)d_in[2];
  (void)d_in[3];                                // W_hh_f: dead (forward state is always zero)
  const float* bihf  = (const float*)d_in[4];
  const float* bhhf  = (const float*)d_in[5];
  const float* Wihb  = (const float*)d_in[6];
  const float* Whhb  = (const float*)d_in[7];
  const float* bihb  = (const float*)d_in[8];
  const float* bhhb  = (const float*)d_in[9];
  float* out = (float*)d_out;

  char* ws = (char*)d_ws;
  size_t off = 0;
  auto alloc = [&](size_t bytes) -> char* {
    char* p = ws + off;
    off = (off + bytes + 255) & ~(size_t)255;
    return p;
  };
  u16* xbf   = (u16*)alloc((size_t)B_ * S_ * EP * 2);   // ~21 MB
  u16* Wf    = (u16*)alloc((size_t)G3 * EP * 2);
  u16* Wb    = (u16*)alloc((size_t)G3 * EP * 2);
  u16* Whh   = (u16*)alloc((size_t)G3 * H_ * 2);
  float* biasF = (float*)alloc(G3 * sizeof(float));
  float* biasB = (float*)alloc(G3 * sizeof(float));
  u16* hping = (u16*)alloc((size_t)B_ * H_ * 2);
  u16* hpong = (u16*)alloc((size_t)B_ * H_ * 2);
  unsigned* bar = (unsigned*)alloc(256);
  u16* xg    = (u16*)alloc((size_t)S_ * G3 * B_ * 2);   // ~101 MB, [S][1536][64]

  init_kernel<<<128, 256, 0, stream>>>(hping, hpong, bar);
  embed_kernel<<<B_ * S_, 64, 0, stream>>>(idx, table, xbf);
  pack_kernel<<<G3, 64, 0, stream>>>(Wihf, Wihb, Whhb, bihf, bhhf, bihb, bhhb,
                                     Wf, Wb, Whh, biasF, biasB);
  fwd_kernel<<<B_ * 4, 256, 0, stream>>>(xbf, Wf, biasF, out);
  bgates_kernel<<<B_ * 4, 256, 0, stream>>>(xbf, Wb, biasB, xg);
  scan_kernel<<<16, 256, 96 * LDS_STRIDE * 2, stream>>>(Whh, xg, hping, hpong, bar, out);
}
